// Attention_final_62491774157449
// MI455X (gfx1250) — compile-verified, matrix operand order checked by reading, not run on hardware
//
#include <hip/hip_runtime.h>

typedef __attribute__((ext_vector_type(16))) _Float16 v16h;
typedef __attribute__((ext_vector_type(8)))  float    v8f;
typedef __attribute__((__vector_size__(4 * sizeof(int)))) int v4i;

#define B_   8
#define N_   4096
#define D_   1024
#define H_   16
#define DH_  64
#define EPSF 1e-5f

// ---------------- workspace layout (bytes) ----------------
#define WS_QN    ((size_t)0)                          // 16*64 f32          = 4 KB
#define WS_WQ    ((size_t)4096)                       // 1024*16 f16 (swz)  = 32 KB
#define WS_XN    ((size_t)65536)                      // 32768*1024 f16     = 64 MB
#define WS_SIM   (WS_XN   + (size_t)B_*N_*D_*2)       // 128*4096 f32       = 2 MB
#define WS_ATTNH (WS_SIM  + (size_t)B_*H_*N_*4)       // 128*4096 f16       = 1 MB
#define WS_Y     (WS_ATTNH+ (size_t)B_*H_*N_*2)       // 8*16*1024 f32      = 512 KB
#define WS_O     (WS_Y    + (size_t)B_*H_*D_*4)       // 8*1024 f32         = 32 KB

// Per-lane K index inside a 16x32 f16 WMMA fragment:
// halves 0..7 -> K 0..7, halves 8..15 -> K 16..23; lanes 16..31 add +8.
__device__ __forceinline__ int kmap(int i, int lane) {
    return (i & 7) + ((i >> 3) << 4) + ((lane >> 4) << 3);
}

union FragU { v16h h; uint4 q[2]; };

// A-fragment: each lane's 16 halves are two contiguous 16B runs -> 2x b128 loads.
__device__ __forceinline__ v16h load_frag_contig(const _Float16* base, int lane) {
    int off = (lane >> 4) << 3;          // 0 or 8 halves
    FragU u;
    u.q[0] = *(const uint4*)(base + off);
    u.q[1] = *(const uint4*)(base + off + 16);
    return u.h;
}

// -------- CDNA5 async global->LDS (direct-to-LDS, tracked by ASYNCcnt) --------
#if defined(__gfx1250__) && __has_builtin(__builtin_amdgcn_global_load_async_to_lds_b128)
#define HAS_ASYNC_LDS 1
#else
#define HAS_ASYNC_LDS 0
#endif

__device__ __forceinline__ void async_copy16(const _Float16* __restrict__ g,
                                             _Float16* __restrict__ l) {
#if HAS_ASYNC_LDS
    __builtin_amdgcn_global_load_async_to_lds_b128(
        (v4i*)const_cast<_Float16*>(g), (v4i*)l, 0, 0);
#else
    ((uint4*)l)[0] = ((const uint4*)g)[0];
#endif
}

__device__ __forceinline__ void wait_async0() {
#if defined(__gfx1250__) && __has_builtin(__builtin_amdgcn_s_wait_asynccnt)
    __builtin_amdgcn_s_wait_asynccnt(0);
#elif defined(__gfx1250__)
    asm volatile("s_wait_asynccnt 0x0" ::: "memory");
#endif
}

// ---------------- 1. LN(q) * scale ----------------
__global__ void k_qn(const float* __restrict__ q, const float* __restrict__ qg,
                     const float* __restrict__ qb, float* __restrict__ qn) {
    int h = threadIdx.x;
    if (h >= H_) return;
    const float* qh = q + h * DH_;
    float s = 0.f, s2 = 0.f;
    #pragma unroll
    for (int j = 0; j < DH_; ++j) { float v = qh[j]; s += v; s2 += v * v; }
    float mu = s * (1.f / DH_);
    float var = s2 * (1.f / DH_) - mu * mu;
    float rstd = rsqrtf(var + EPSF);
    const float scale = 0.125f;  // 64^-0.5
    #pragma unroll
    for (int j = 0; j < DH_; ++j)
        qn[h * DH_ + j] = ((qh[j] - mu) * rstd * qg[j] + qb[j]) * scale;
}

// ------- 2. wq_sw: Wk_h @ qn_h, stored pre-swizzled in B-fragment order -------
// Fragment order: wq_sw[(g*32 + lane)*16 + i] where lane/i invert kmap for k=g*32+l.
__global__ __launch_bounds__(256) void k_fold(const float* __restrict__ Wkv,
                                              const float* __restrict__ qn,
                                              _Float16* __restrict__ wq_sw) {
    int idx = blockIdx.x * 256 + threadIdx.x;    // 16384 = 1024 * 16
    int d = idx >> 4, h = idx & 15;
    const float* w  = Wkv + (size_t)d * (2 * D_) + h * DH_;
    const float* qh = qn + h * DH_;
    float acc = 0.f;
    #pragma unroll 8
    for (int j = 0; j < DH_; ++j) acc += w[j] * qh[j];
    int g   = d >> 5;
    int l   = d & 31;
    int lam = (l >> 3) & 1;                      // lane group (0: lanes 0-15, 1: 16-31)
    int i   = (l & 7) + (((l >> 4) & 1) << 3);   // half index within fragment
    wq_sw[((g * 32) + h + (lam << 4)) * 16 + i] = (_Float16)acc;
}

// ---------------- 3. LN(x) tile -> LDS/f16 ws, then sim = xn @ wq via WMMA ----------------
__global__ __launch_bounds__(256) void k_simln(const float* __restrict__ x,
                                               const float* __restrict__ ng,
                                               const float* __restrict__ nb,
                                               const _Float16* __restrict__ wq_sw,
                                               _Float16* __restrict__ xn,
                                               float* __restrict__ sim) {
    __shared__ __align__(16) _Float16 tile[16 * D_];   // 32 KB
    int b    = blockIdx.x >> 8;        // N/16 = 256 tiles per batch
    int t    = blockIdx.x & 255;
    int s0   = t << 4;
    int wave = threadIdx.x >> 5;
    int lane = threadIdx.x & 31;

    // 8 waves x 2 rows: layernorm 16 rows, spill f16 to LDS + global ws
    for (int rr = 0; rr < 2; ++rr) {
        int r = wave * 2 + rr;
        size_t row = (size_t)(b * N_ + s0 + r);
        const float* xr = x + row * D_;
        if (rr == 0) __builtin_prefetch(xr + D_, 0, 0);   // global_prefetch next row
        float v[32], s = 0.f, s2 = 0.f;
        #pragma unroll
        for (int i = 0; i < 32; ++i) {
            v[i] = xr[lane + (i << 5)];
            s += v[i]; s2 += v[i] * v[i];
        }
        #pragma unroll
        for (int off = 16; off; off >>= 1) {
            s  += __shfl_xor(s,  off, 32);
            s2 += __shfl_xor(s2, off, 32);
        }
        float mu   = s * (1.f / D_);
        float var  = s2 * (1.f / D_) - mu * mu;
        float rstd = rsqrtf(var + EPSF);
        _Float16* dl = tile + r * D_;
        _Float16* dg = xn + row * D_;
        #pragma unroll
        for (int i = 0; i < 32; ++i) {
            int c = lane + (i << 5);
            _Float16 hv = (_Float16)((v[i] - mu) * rstd * ng[c] + nb[c]);
            dl[c] = hv;
            dg[c] = hv;
        }
    }
    __syncthreads();

    // wave 0: 16x16 sim tile = tile[16x1024] @ wq[1024x16], dual-accumulator WMMA
    if (wave == 0) {
        v8f acc0 = {}, acc1 = {};
        int rA = lane & 15;   // A row (token in tile)
        int nB = lane & 15;   // B col (head)
        const _Float16* arow = tile + rA * D_;
        for (int k0 = 0; k0 < D_; k0 += 64) {
            v16h a0 = load_frag_contig(arow + k0, lane);
            v16h a1 = load_frag_contig(arow + k0 + 32, lane);
            // pre-swizzled B fragments: 32B contiguous per lane, coalesced
            FragU ub0, ub1;
            const uint4* bp0 = (const uint4*)(wq_sw + ((size_t)((k0 >> 5) * 32 + lane)) * 16);
            const uint4* bp1 = (const uint4*)(wq_sw + ((size_t)(((k0 >> 5) + 1) * 32 + lane)) * 16);
            ub0.q[0] = bp0[0]; ub0.q[1] = bp0[1];
            ub1.q[0] = bp1[0]; ub1.q[1] = bp1[1];
            acc0 = __builtin_amdgcn_wmma_f32_16x16x32_f16(
                false, a0, false, ub0.h, (short)0, acc0, false, false);
            acc1 = __builtin_amdgcn_wmma_f32_16x16x32_f16(
                false, a1, false, ub1.h, (short)0, acc1, false, false);
        }
        v8f acc = acc0 + acc1;
        int mbase = (lane < 16) ? 0 : 8;
        #pragma unroll
        for (int i = 0; i < 8; ++i) {
            int srow = s0 + mbase + i;
            sim[(size_t)(b * H_ + nB) * N_ + srow] = acc[i];
        }
    }
}

// ---------------- 4. softmax over n=4096 per (b,h) ----------------
__global__ __launch_bounds__(256) void k_softmax(const float* __restrict__ sim,
                                                 float* __restrict__ attn,
                                                 _Float16* __restrict__ attnh) {
    __shared__ float red[8];
    size_t row = blockIdx.x;           // b*16 + h
    const float* sr = sim + row * N_;
    int tid = threadIdx.x, lane = tid & 31, wave = tid >> 5;
    float vals[16], m = -1e30f;
    #pragma unroll
    for (int i = 0; i < 16; ++i) { vals[i] = sr[tid + (i << 8)]; m = fmaxf(m, vals[i]); }
    #pragma unroll
    for (int off = 16; off; off >>= 1) m = fmaxf(m, __shfl_xor(m, off, 32));
    if (lane == 0) red[wave] = m;
    __syncthreads();
    m = red[0];
    #pragma unroll
    for (int i = 1; i < 8; ++i) m = fmaxf(m, red[i]);
    float s = 0.f;
    #pragma unroll
    for (int i = 0; i < 16; ++i) { vals[i] = __expf(vals[i] - m); s += vals[i]; }
    #pragma unroll
    for (int off = 16; off; off >>= 1) s += __shfl_xor(s, off, 32);
    __syncthreads();
    if (lane == 0) red[wave] = s;
    __syncthreads();
    s = 0.f;
    #pragma unroll
    for (int i = 0; i < 8; ++i) s += red[i];
    float inv = 1.f / s;
    #pragma unroll
    for (int i = 0; i < 16; ++i) {
        float a = vals[i] * inv;
        attn [row * N_ + tid + (i << 8)] = a;
        attnh[row * N_ + tid + (i << 8)] = (_Float16)a;
    }
}

// ---------------- 5. y[b,h,:] = attn[b,h,:] @ xn[b,:,:]  (16 x 4096 x 1024) ----
// Double-buffered: async global->LDS for tile k+1 overlaps WMMA on tile k.
__global__ __launch_bounds__(256) void k_y(const _Float16* __restrict__ attnh,
                                           const _Float16* __restrict__ xn,
                                           float* __restrict__ y) {
    __shared__ __align__(16) _Float16 bt[2][32 * 128];  // 2 x 8 KB B tiles (K x Ncols)
    __shared__ __align__(16) _Float16 at[2][16 * 32];   // 2 x 1 KB A tiles (heads x K)
    int b  = blockIdx.x >> 3;
    int n0 = (blockIdx.x & 7) << 7;                     // 128-col slab
    int tid = threadIdx.x, lane = tid & 31, wave = tid >> 5;

    // staging coordinates (16 contiguous halves per thread)
    int i0 = tid << 4;
    int kr = i0 >> 7, nc = i0 & 127;
    const _Float16* bsrc = xn + ((size_t)(b * N_ + kr) * D_ + n0 + nc);
    int hr = (tid << 4) >> 5, kk = (tid << 4) & 31;     // for tid < 32
    const _Float16* asrc = attnh + (size_t)(b * H_ + hr) * N_ + kk;

    auto stage = [&](int k0, int buf) {
        async_copy16(bsrc + (size_t)k0 * D_, &bt[buf][kr * 128 + nc]);
        if (tid < 32)
            async_copy16(asrc + k0, &at[buf][hr * 32 + kk]);
    };

    stage(0, 0);
    wait_async0();
    __syncthreads();

    v8f acc0 = {}, acc1 = {};
    int rA   = lane & 15;               // A row = head
    int colB = (wave << 4) + (lane & 15);
    int buf = 0;
    for (int k0 = 0; k0 < N_; k0 += 32) {
        int nbuf = buf ^ 1;
        if (k0 + 32 < N_) stage(k0 + 32, nbuf);   // async fill next tile
        v16h a = load_frag_contig(&at[buf][rA * 32], lane);   // 2x ds_load_b128
        v16h bb;
        #pragma unroll
        for (int i = 0; i < 16; ++i) {            // B: k-stride 256B, true gather
            int k = kmap(i, lane);
            bb[i] = bt[buf][k * 128 + colB];
        }
        if ((k0 >> 5) & 1)
            acc1 = __builtin_amdgcn_wmma_f32_16x16x32_f16(
                false, a, false, bb, (short)0, acc1, false, false);
        else
            acc0 = __builtin_amdgcn_wmma_f32_16x16x32_f16(
                false, a, false, bb, (short)0, acc0, false, false);
        wait_async0();      // next tile landed in LDS
        __syncthreads();    // all waves done reading buf / writing nbuf
        buf = nbuf;
    }
    v8f acc = acc0 + acc1;
    int mbase = (lane < 16) ? 0 : 8;
    #pragma unroll
    for (int i = 0; i < 8; ++i) {
        int h = mbase + i;
        y[(size_t)(b * H_ + h) * D_ + n0 + colB] = acc[i];
    }
}

// ---------------- 6. o[b][c] = y[b, c/64, :] . Wv[:, c] + bv[c] ----------------
__global__ __launch_bounds__(256) void k_o(const float* __restrict__ y,
                                           const float* __restrict__ Wkv,
                                           const float* __restrict__ bkv,
                                           float* __restrict__ o) {
    int idx = blockIdx.x * 256 + threadIdx.x;   // 8192
    int b = idx >> 10, c = idx & 1023;
    int h = c >> 6;
    const float* yr = y + (size_t)(b * H_ + h) * D_;
    float acc = bkv[D_ + c];
    #pragma unroll 4
    for (int d = 0; d < D_; ++d)
        acc += yr[d] * Wkv[(size_t)d * (2 * D_) + D_ + c];
    o[idx] = acc;
}

// ---------------- 7. out[b][d] = o[b,:] . Wout[:,d] + bout[d] ----------------
__global__ __launch_bounds__(256) void k_out(const float* __restrict__ o,
                                             const float* __restrict__ Wout,
                                             const float* __restrict__ bout,
                                             float* __restrict__ out) {
    int idx = blockIdx.x * 256 + threadIdx.x;   // 8192
    int b = idx >> 10, d = idx & 1023;
    const float* ob = o + b * D_;
    float acc = bout[d];
    #pragma unroll 4
    for (int i = 0; i < D_; ++i)
        acc += ob[i] * Wout[(size_t)i * D_ + d];
    out[idx] = acc;
}

extern "C" void kernel_launch(void* const* d_in, const int* in_sizes, int n_in,
                              void* d_out, int out_size, void* d_ws, size_t ws_size,
                              hipStream_t stream) {
    (void)in_sizes; (void)n_in; (void)out_size; (void)ws_size;
    const float* x    = (const float*)d_in[0];
    const float* q    = (const float*)d_in[1];
    const float* Wkv  = (const float*)d_in[2];
    const float* bkv  = (const float*)d_in[3];
    const float* Wout = (const float*)d_in[4];
    const float* bout = (const float*)d_in[5];
    const float* ng   = (const float*)d_in[6];
    const float* nb   = (const float*)d_in[7];
    const float* qg   = (const float*)d_in[8];
    const float* qb   = (const float*)d_in[9];

    char* ws = (char*)d_ws;
    float*    qn    = (float*)(ws + WS_QN);
    _Float16* wq_sw = (_Float16*)(ws + WS_WQ);
    _Float16* xn    = (_Float16*)(ws + WS_XN);
    float*    sim   = (float*)(ws + WS_SIM);
    _Float16* attnh = (_Float16*)(ws + WS_ATTNH);
    float*    y     = (float*)(ws + WS_Y);
    float*    o     = (float*)(ws + WS_O);

    float* out  = (float*)d_out;                 // [8, 1024]
    float* attn = (float*)d_out + B_ * D_;       // [8, 16, 1, 4096]

    k_qn     <<<1, 64, 0, stream>>>(q, qg, qb, qn);
    k_fold   <<<64, 256, 0, stream>>>(Wkv, qn, wq_sw);
    k_simln  <<<B_ * (N_ / 16), 256, 0, stream>>>(x, ng, nb, wq_sw, xn, sim);
    k_softmax<<<B_ * H_, 256, 0, stream>>>(sim, attn, attnh);
    k_y      <<<B_ * 8, 256, 0, stream>>>(attnh, xn, y);
    k_o      <<<32, 256, 0, stream>>>(y, Wkv, bkv, o);
    k_out    <<<32, 256, 0, stream>>>(o, Wout, bout, out);
}
